// DeepSeekMLA_42631845380533
// MI455X (gfx1250) — matrix operand order repaired and checked by reading, not compiled, not run on hardware
//
#include <hip/hip_runtime.h>
#include <hip/hip_bf16.h>

// ---------------------------------------------------------------------------
// DeepSeek MLA forward for gfx1250 (MI455X), bf16 WMMA pipeline + TDM staging.
// B=2 S=2048 HID=2048 H=16 DN=128 DR=64 DV=128 R=512 QHD=192
// ---------------------------------------------------------------------------

#define Bc 2
#define Sc 2048
#define HIDc 2048
#define Hc 16
#define DNc 128
#define DRc 64
#define DVc 128
#define Rc 512
#define QHDc 192
#define Mrows (Bc * Sc)   // 4096

typedef __attribute__((ext_vector_type(16))) __bf16 v16bf;
typedef __attribute__((ext_vector_type(8)))  float  v8f;
typedef __attribute__((ext_vector_type(4)))  unsigned int v4u;
typedef __attribute__((ext_vector_type(8)))  int v8i;
typedef __attribute__((ext_vector_type(4)))  int v4i;

union BF16x16 { v16bf v; unsigned short u[16]; };

__device__ __forceinline__ unsigned short f2bf(float f) {
  unsigned int u = __float_as_uint(f);
  unsigned int r = u + 0x7FFFu + ((u >> 16) & 1u);   // round-to-nearest-even
  return (unsigned short)(r >> 16);
}

// A-fragment (16-bit, 16xK tile, K-chunk of 32): lane = row (lane&15),
// g = lane>>4 selects K halves: j<8 -> K=g*8+j ; j>=8 -> K=16+g*8+(j-8).
__device__ __forceinline__ v16bf load_a_frag(const unsigned short* p, int ld, int k0) {
  int lane = threadIdx.x & 31;
  int g = lane >> 4, m = lane & 15;
  const unsigned short* q = p + (size_t)m * ld + k0;
  BF16x16 r;
#pragma unroll
  for (int j = 0; j < 8; ++j) r.u[j] = q[g * 8 + j];
#pragma unroll
  for (int j = 0; j < 8; ++j) r.u[8 + j] = q[16 + g * 8 + j];
  return r.v;
}

// B-fragment (16-bit, Kx16 tile, K-chunk of 32): lane = column (lane&15),
// j<8 -> K=(lane>>4)*8+j ; j>=8 -> K=16+(lane>>4)*8+(j-8).  Source stored row-major [N, K].
__device__ __forceinline__ v16bf load_b_frag(const unsigned short* p, int ld, int k0) {
  int lane = threadIdx.x & 31;
  int g = lane >> 4, n = lane & 15;
  const unsigned short* q = p + (size_t)n * ld + k0;
  BF16x16 r;
#pragma unroll
  for (int j = 0; j < 8; ++j) r.u[j] = q[g * 8 + j];
#pragma unroll
  for (int j = 0; j < 8; ++j) r.u[8 + j] = q[16 + g * 8 + j];
  return r.v;
}

// ---------------------------------------------------------------------------
// Tensor Data Mover: async-load a 2D bf16 tile [TILE_Y rows x TILE_X elems]
// from a row-major tensor (row stride = STRIDE elems, TENSOR_Y rows total)
// into LDS at lds_off.  D# layout per CDNA5 ISA ch.8 (group0/group1).
// ---------------------------------------------------------------------------
template <int TILE_X, int TILE_Y, int TENSOR_X, int TENSOR_Y, int STRIDE>
__device__ __forceinline__ void tdm_load_tile_bf16(unsigned lds_off,
                                                   const unsigned short* gptr) {
  unsigned long long ga = (unsigned long long)gptr;
  unsigned ga_lo = (unsigned)__builtin_amdgcn_readfirstlane((int)(ga & 0xffffffffull));
  unsigned ga_hi = (unsigned)__builtin_amdgcn_readfirstlane((int)(ga >> 32));
  unsigned loff  = (unsigned)__builtin_amdgcn_readfirstlane((int)lds_off);

  v4u g0;
  g0.x = 1u;                                    // count=1 (valid), user mode
  g0.y = loff;                                  // lds_addr [63:32]
  g0.z = ga_lo;                                 // global_addr [95:64]
  g0.w = (ga_hi & 0x1ffffffu) | (2u << 30);     // global_addr[56:32] | type=2

  v8i g1;
  g1[0] = (int)(1u << 16);                                  // data_size=1 (2B), wg_mask=0
  g1[1] = (int)((unsigned)TENSOR_X << 16);                  // tensor_dim0[15:0] @ [63:48]
  g1[2] = (int)((((unsigned)TENSOR_X) >> 16) |
                ((unsigned)TENSOR_Y << 16));                // dim0 hi | tensor_dim1 lo
  g1[3] = (int)((((unsigned)TENSOR_Y) >> 16) |
                ((unsigned)TILE_X << 16));                  // dim1 hi | tile_dim0
  g1[4] = (int)((unsigned)TILE_Y);                          // tile_dim1 | tile_dim2=0
  g1[5] = (int)((unsigned)STRIDE);                          // tensor_dim0_stride lo32
  g1[6] = 0;                                                // stride hi | dim1_stride lo
  g1[7] = 0;

  v4i z4 = {0, 0, 0, 0};
#if defined(__clang_major__) && (__clang_major__ >= 23)
  v8i z8 = {0, 0, 0, 0, 0, 0, 0, 0};
  __builtin_amdgcn_tensor_load_to_lds(g0, g1, z4, z4, z8, 0);
#else
  __builtin_amdgcn_tensor_load_to_lds(g0, g1, z4, z4, 0);
#endif
}

// ---------------------------------------------------------------------------
// fp32 -> bf16 conversion (grid-stride)
// ---------------------------------------------------------------------------
__global__ void cvt_f32_bf16_kernel(const float* __restrict__ src,
                                    unsigned short* __restrict__ dst, size_t n) {
  size_t i = (size_t)blockIdx.x * blockDim.x + threadIdx.x;
  size_t stride = (size_t)gridDim.x * blockDim.x;
  for (; i < n; i += stride) dst[i] = f2bf(src[i]);
}

// ---------------------------------------------------------------------------
// C[M,N] = A[M,K] * W[N,K]^T   (bf16 in, fp32 out)
// One wave computes a 16x64 output tile. Software-pipelined: fragments for
// k+32 are issued before the WMMAs for k consume their data, so the compiler
// can emit partial s_wait_loadcnt and keep loads in flight across the MMAs.
// ---------------------------------------------------------------------------
__global__ __launch_bounds__(256) void gemm_bf16_kernel(
    const unsigned short* __restrict__ A, const unsigned short* __restrict__ W,
    float* __restrict__ C, int M, int N, int K) {
  int wid = (int)(((size_t)blockIdx.x * blockDim.x + threadIdx.x) >> 5);
  int n4 = N >> 6;
  int mt = wid / n4, nt = wid % n4;
  if (mt * 16 >= M) return;
  int lane = threadIdx.x & 31;
  const unsigned short* Arow = A + (size_t)mt * 16 * K;
  const unsigned short* W0 = W + (size_t)(nt * 64 +  0) * K;
  const unsigned short* W1 = W + (size_t)(nt * 64 + 16) * K;
  const unsigned short* W2 = W + (size_t)(nt * 64 + 32) * K;
  const unsigned short* W3 = W + (size_t)(nt * 64 + 48) * K;

  v8f acc0 = {}, acc1 = {}, acc2 = {}, acc3 = {};
  // prologue
  v16bf a  = load_a_frag(Arow, K, 0);
  v16bf b0 = load_b_frag(W0, K, 0);
  v16bf b1 = load_b_frag(W1, K, 0);
  v16bf b2 = load_b_frag(W2, K, 0);
  v16bf b3 = load_b_frag(W3, K, 0);
  for (int k0 = 0; k0 < K; k0 += 32) {
    v16bf an = a, bn0 = b0, bn1 = b1, bn2 = b2, bn3 = b3;
    if (k0 + 32 < K) {   // issue next k-chunk loads before consuming current
      __builtin_prefetch(Arow + (size_t)(lane & 15) * K + k0 + 64, 0, 1);
      an  = load_a_frag(Arow, K, k0 + 32);
      bn0 = load_b_frag(W0, K, k0 + 32);
      bn1 = load_b_frag(W1, K, k0 + 32);
      bn2 = load_b_frag(W2, K, k0 + 32);
      bn3 = load_b_frag(W3, K, k0 + 32);
    }
    acc0 = __builtin_amdgcn_wmma_f32_16x16x32_bf16(false, a, false, b0, (short)0, acc0, false, false);
    acc1 = __builtin_amdgcn_wmma_f32_16x16x32_bf16(false, a, false, b1, (short)0, acc1, false, false);
    acc2 = __builtin_amdgcn_wmma_f32_16x16x32_bf16(false, a, false, b2, (short)0, acc2, false, false);
    acc3 = __builtin_amdgcn_wmma_f32_16x16x32_bf16(false, a, false, b3, (short)0, acc3, false, false);
    a = an; b0 = bn0; b1 = bn1; b2 = bn2; b3 = bn3;
  }
  int g = lane >> 4, n = lane & 15;
#pragma unroll
  for (int i = 0; i < 8; ++i) {
    int row = mt * 16 + i + 8 * g;
    float* crow = C + (size_t)row * N + nt * 64 + n;
    crow[0] = acc0[i]; crow[16] = acc1[i]; crow[32] = acc2[i]; crow[48] = acc3[i];
  }
}

// ---------------------------------------------------------------------------
// RMSNorm over R=512, one wave per row, output bf16
// ---------------------------------------------------------------------------
__global__ __launch_bounds__(256) void rmsnorm_kernel(
    const float* __restrict__ ckv, const float* __restrict__ w,
    unsigned short* __restrict__ out) {
  int row = (int)(((size_t)blockIdx.x * blockDim.x + threadIdx.x) >> 5);
  int lane = threadIdx.x & 31;
  if (row >= Mrows) return;
  const float* x = ckv + (size_t)row * Rc;
  float vals[16];
  float ss = 0.f;
#pragma unroll
  for (int j = 0; j < 16; ++j) { float t = x[lane * 16 + j]; vals[j] = t; ss += t * t; }
#pragma unroll
  for (int off = 16; off >= 1; off >>= 1) ss += __shfl_xor(ss, off, 32);
  float scale = rsqrtf(ss * (1.0f / (float)Rc) + 1e-5f);
#pragma unroll
  for (int j = 0; j < 16; ++j)
    out[(size_t)row * Rc + lane * 16 + j] = f2bf(vals[j] * scale * w[lane * 16 + j]);
}

// ---------------------------------------------------------------------------
// RoPE + repack:
//   qf/kf fp32 [B*S, H*QHD] -> Qb/Kb bf16 [B,H,S,QHD] (rope on dims DN..DN+DR)
//   vf fp32 [B*S, H*DV]     -> Vt bf16 [B,H,DV,S]     (transposed for B-frags)
// ---------------------------------------------------------------------------
__global__ __launch_bounds__(256) void rope_pack_kernel(
    const float* __restrict__ qf, const float* __restrict__ kf,
    const float* __restrict__ vf, const float* __restrict__ cosp,
    const float* __restrict__ sinp, unsigned short* __restrict__ Qb,
    unsigned short* __restrict__ Kb, unsigned short* __restrict__ Vt) {
  int sg = blockIdx.x;            // 0 .. B*S-1
  int b = sg >> 11, s = sg & (Sc - 1);
  const float* cr = cosp + (size_t)s * DRc;
  const float* sr = sinp + (size_t)s * DRc;
  const float* qrow = qf + (size_t)sg * (Hc * QHDc);
  const float* krow = kf + (size_t)sg * (Hc * QHDc);
  for (int idx = threadIdx.x; idx < Hc * QHDc; idx += blockDim.x) {
    int h = idx / QHDc, hd = idx % QHDc;
    float qv = qrow[idx], kv = krow[idx];
    if (hd >= DNc) {
      int r = hd - DNc;
      float c = cr[r], sn = sr[r];
      if (r < DRc / 2) {
        qv = qv * c - qrow[idx + DRc / 2] * sn;
        kv = kv * c - krow[idx + DRc / 2] * sn;
      } else {
        qv = qv * c + qrow[idx - DRc / 2] * sn;
        kv = kv * c + krow[idx - DRc / 2] * sn;
      }
    }
    size_t o = ((size_t)(b * Hc + h) * Sc + s) * QHDc + hd;
    Qb[o] = f2bf(qv);
    Kb[o] = f2bf(kv);
  }
  const float* vrow = vf + (size_t)sg * (Hc * DVc);
  for (int idx = threadIdx.x; idx < Hc * DVc; idx += blockDim.x) {
    int h = idx >> 7, d = idx & (DVc - 1);
    Vt[((size_t)(b * Hc + h) * DVc + d) * Sc + s] = f2bf(vrow[idx]);
  }
}

// ---------------------------------------------------------------------------
// Flash attention: one wave per (b, h, 16-row q-tile). K tiles (32 x QHD bf16)
// are staged into LDS by the Tensor Data Mover, double-buffered: the TDM for
// tile kb+1 runs while the WMMAs consume tile kb (s_wait_tensorcnt 1/0).
// Online softmax in fp32; P tile staged through LDS for the P.V WMMA.
// 2 waves / 64 threads per block: LDS = 2*2*12KB (K tiles) + 2KB (P) = 50KB.
// ---------------------------------------------------------------------------
__global__ __launch_bounds__(64) void mla_attn_kernel(
    const unsigned short* __restrict__ Qb, const unsigned short* __restrict__ Kb,
    const unsigned short* __restrict__ Vt, unsigned short* __restrict__ attnb) {
  __shared__ unsigned short kbuf[2][2][32 * QHDc];  // [wave][buf][32 keys x 192]
  __shared__ unsigned short pbuf[2][16 * 32];       // per-wave P tile (16q x 32k)

  int wid = (int)(((size_t)blockIdx.x * blockDim.x + threadIdx.x) >> 5);
  int wv = threadIdx.x >> 5;
  int lane = threadIdx.x & 31;
  int g = lane >> 4, ln = lane & 15;

  int qt = wid & 127;          // S/16 = 128 q tiles
  int h  = (wid >> 7) & (Hc - 1);
  int b  = wid >> 11;
  int qbase = qt * 16;

  const unsigned short* Qh = Qb + (size_t)(b * Hc + h) * Sc * QHDc;
  const unsigned short* Kh = Kb + (size_t)(b * Hc + h) * Sc * QHDc;
  const unsigned short* Vh = Vt + (size_t)(b * Hc + h) * DVc * Sc;

  v16bf qfrag[6];
#pragma unroll
  for (int c = 0; c < 6; ++c)
    qfrag[c] = load_a_frag(Qh + (size_t)qbase * QHDc, QHDc, c * 32);

  v8f zero = {};
  v8f acc[8];
#pragma unroll
  for (int t = 0; t < 8; ++t) acc[t] = zero;
  float mrow[8], lrow[8];
#pragma unroll
  for (int i = 0; i < 8; ++i) { mrow[i] = -3.0e38f; lrow[i] = 0.0f; }

  const float scale = 0.07216878364870323f;   // 1/sqrt(QHD=192)
  int kbmax = (qbase + 15) >> 5;

  // TDM prologue: stage K tile 0 into buffer 0
  tdm_load_tile_bf16<QHDc, 32, QHDc, Sc, QHDc>(
      (unsigned)(unsigned long long)(const void*)&kbuf[wv][0][0], Kh);

  for (int kb = 0; kb <= kbmax; ++kb) {
    int key0 = kb * 32;
    // Issue TDM for the next K tile into the other buffer, then wait so the
    // current buffer is ready while the next transfer stays in flight.
    if (kb < kbmax) {
      tdm_load_tile_bf16<QHDc, 32, QHDc, Sc, QHDc>(
          (unsigned)(unsigned long long)(const void*)&kbuf[wv][(kb + 1) & 1][0],
          Kh + (size_t)(key0 + 32) * QHDc);
      __builtin_amdgcn_s_wait_tensorcnt(1);
    } else {
      __builtin_amdgcn_s_wait_tensorcnt(0);
    }
    asm volatile("" ::: "memory");
    const unsigned short* kt = &kbuf[wv][kb & 1][0];   // [32 x 192] row-major

    // --- scores: two 16-key tiles from LDS, K=QHD in 6 chunks of 32 ---
    v8f s0 = zero, s1 = zero;
#pragma unroll
    for (int c = 0; c < 6; ++c) {
      v16bf b0 = load_b_frag(kt, QHDc, c * 32);
      v16bf b1 = load_b_frag(kt + 16 * QHDc, QHDc, c * 32);
      s0 = __builtin_amdgcn_wmma_f32_16x16x32_bf16(false, qfrag[c], false, b0, (short)0, s0, false, false);
      s1 = __builtin_amdgcn_wmma_f32_16x16x32_bf16(false, qfrag[c], false, b1, (short)0, s1, false, false);
    }
    // --- online softmax over the 32 keys, per row ---
#pragma unroll
    for (int i = 0; i < 8; ++i) {
      int row = qbase + i + 8 * g;
      float a0 = (key0 + ln      <= row) ? s0[i] * scale : -1.0e9f;
      float a1 = (key0 + 16 + ln <= row) ? s1[i] * scale : -1.0e9f;
      float tmax = fmaxf(a0, a1);
#pragma unroll
      for (int off = 8; off >= 1; off >>= 1) tmax = fmaxf(tmax, __shfl_xor(tmax, off, 16));
      float mnew = fmaxf(mrow[i], tmax);
      float f  = __expf(mrow[i] - mnew);
      float p0 = __expf(a0 - mnew);
      float p1 = __expf(a1 - mnew);
      float rs = p0 + p1;
#pragma unroll
      for (int off = 8; off >= 1; off >>= 1) rs += __shfl_xor(rs, off, 16);
      lrow[i] = lrow[i] * f + rs;
      mrow[i] = mnew;
#pragma unroll
      for (int t = 0; t < 8; ++t) acc[t][i] *= f;
      int prow = i + 8 * g;
      pbuf[wv][prow * 32 + ln]      = f2bf(p0);
      pbuf[wv][prow * 32 + 16 + ln] = f2bf(p1);
    }
    asm volatile("s_wait_dscnt 0" ::: "memory");   // cross-lane LDS RAW on P
    // --- reload P as A-fragment (16x32, ld=32) ---
    BF16x16 pa;
    {
      const unsigned short* pp = &pbuf[wv][0];
#pragma unroll
      for (int j = 0; j < 8; ++j) pa.u[j]     = pp[ln * 32 + g * 8 + j];
#pragma unroll
      for (int j = 0; j < 8; ++j) pa.u[8 + j] = pp[ln * 32 + 16 + g * 8 + j];
    }
    // --- P.V accumulate over DV in 8 column tiles (V from global, bf16) ---
#pragma unroll
    for (int t = 0; t < 8; ++t) {
      v16bf vb = load_b_frag(Vh + (size_t)(t * 16) * Sc, Sc, key0);
      acc[t] = __builtin_amdgcn_wmma_f32_16x16x32_bf16(false, pa.v, false, vb, (short)0, acc[t], false, false);
    }
  }
  // --- finalize: divide by l, write bf16 [B, S, H*DV] for O-proj ---
#pragma unroll
  for (int i = 0; i < 8; ++i) {
    float inv = 1.0f / lrow[i];
    int row = qbase + i + 8 * g;
    size_t base = ((size_t)(b * Sc + row)) * (Hc * DVc) + h * DVc + ln;
#pragma unroll
    for (int t = 0; t < 8; ++t)
      attnb[base + t * 16] = f2bf(acc[t][i] * inv);
  }
}

// ---------------------------------------------------------------------------
// Host orchestration
// ---------------------------------------------------------------------------
extern "C" void kernel_launch(void* const* d_in, const int* in_sizes, int n_in,
                              void* d_out, int out_size, void* d_ws, size_t ws_size,
                              hipStream_t stream) {
  (void)in_sizes; (void)n_in; (void)out_size; (void)ws_size;
  const float* hidden   = (const float*)d_in[0];
  // d_in[1]=attention_mask (causal, applied analytically), d_in[2]=position_ids (== arange)
  const float* Wq       = (const float*)d_in[3];
  const float* Wkv_down = (const float*)d_in[4];
  const float* kv_nw    = (const float*)d_in[5];
  const float* Wkv_up   = (const float*)d_in[6];
  const float* Wkv_v    = (const float*)d_in[7];
  const float* Wo       = (const float*)d_in[8];
  const float* cosp     = (const float*)d_in[9];
  const float* sinp     = (const float*)d_in[10];
  float* out = (float*)d_out;

  char* ws = (char*)d_ws;
  size_t off = 0;
  auto carve = [&](size_t bytes) -> char* {
    char* p = ws + off;
    off += (bytes + 255) & ~(size_t)255;
    return p;
  };
  unsigned short* hb    = (unsigned short*)carve((size_t)Mrows * HIDc * 2);
  unsigned short* wqb   = (unsigned short*)carve((size_t)Hc * QHDc * HIDc * 2);
  unsigned short* wdnb  = (unsigned short*)carve((size_t)Rc * HIDc * 2);
  unsigned short* wupb  = (unsigned short*)carve((size_t)Hc * QHDc * Rc * 2);
  unsigned short* wvb   = (unsigned short*)carve((size_t)Hc * DVc * Rc * 2);
  unsigned short* wob   = (unsigned short*)carve((size_t)HIDc * Hc * DVc * 2);
  float*          qf    = (float*)carve((size_t)Mrows * Hc * QHDc * 4);
  float*          ckv   = (float*)carve((size_t)Mrows * Rc * 4);
  unsigned short* ckvnb = (unsigned short*)carve((size_t)Mrows * Rc * 2);
  float*          kf    = (float*)carve((size_t)Mrows * Hc * QHDc * 4);
  float*          vf    = (float*)carve((size_t)Mrows * Hc * DVc * 4);
  unsigned short* Qb    = (unsigned short*)carve((size_t)Mrows * Hc * QHDc * 2);
  unsigned short* Kb    = (unsigned short*)carve((size_t)Mrows * Hc * QHDc * 2);
  unsigned short* Vtb   = (unsigned short*)carve((size_t)Mrows * Hc * DVc * 2);
  unsigned short* attnb = (unsigned short*)carve((size_t)Mrows * Hc * DVc * 2);

  auto cvt = [&](const float* s, unsigned short* d, size_t n) {
    cvt_f32_bf16_kernel<<<2048, 256, 0, stream>>>(s, d, n);
  };
  cvt(hidden,   hb,   (size_t)Mrows * HIDc);
  cvt(Wq,       wqb,  (size_t)Hc * QHDc * HIDc);
  cvt(Wkv_down, wdnb, (size_t)Rc * HIDc);
  cvt(Wkv_up,   wupb, (size_t)Hc * QHDc * Rc);
  cvt(Wkv_v,    wvb,  (size_t)Hc * DVc * Rc);
  cvt(Wo,       wob,  (size_t)HIDc * Hc * DVc);

  auto gemm = [&](const unsigned short* A, const unsigned short* W, float* C,
                  int M, int N, int K) {
    int waves = (M / 16) * (N / 64);
    int blocks = (waves * 32 + 255) / 256;
    gemm_bf16_kernel<<<blocks, 256, 0, stream>>>(A, W, C, M, N, K);
  };

  // Projections
  gemm(hb, wqb,  qf,  Mrows, Hc * QHDc, HIDc);   // q = hidden @ Wq^T
  gemm(hb, wdnb, ckv, Mrows, Rc,        HIDc);   // c_kv = hidden @ Wkv_down^T
  rmsnorm_kernel<<<Mrows / 8, 256, 0, stream>>>(ckv, kv_nw, ckvnb);
  gemm(ckvnb, wupb, kf, Mrows, Hc * QHDc, Rc);   // k = c_kv @ Wkv_up^T
  gemm(ckvnb, wvb,  vf, Mrows, Hc * DVc,  Rc);   // v = c_kv @ Wkv_v^T

  // RoPE + repack to attention-friendly bf16 layouts
  rope_pack_kernel<<<Mrows, 256, 0, stream>>>(qf, kf, vf, cosp, sinp, Qb, Kb, Vtb);

  // Flash attention: B*H*(S/16) = 4096 waves, 2 waves per block
  mla_attn_kernel<<<(Bc * Hc * (Sc / 16) * 32) / 64, 64, 0, stream>>>(Qb, Kb, Vtb, attnb);

  // Output projection to fp32 d_out
  gemm(attnb, wob, out, Mrows, HIDc, Hc * DVc);
}
